// KL_PS_Loss_85864986181898
// MI455X (gfx1250) — compile-verified, matrix operand order checked by reading, not run on hardware
//
#include <hip/hip_runtime.h>
#include <hip/hip_bf16.h>
#include <cstdint>

typedef __attribute__((ext_vector_type(16))) _Float16 v16h;
typedef __attribute__((ext_vector_type(8)))  _Float16 v8h;
typedef __attribute__((ext_vector_type(8)))  float    v8f;

union Frag { v16h v; v8h h[2]; };

#define N_ROWS   512
#define DIMS     128
#define N_MC     1000
#define SCHUNKS  8
#define SPERCH   (N_MC / SCHUNKS)   // 125

// ---------------- deterministic counter hash RNG ----------------
__device__ __forceinline__ uint32_t hashu(uint32_t x) {
  x ^= x >> 16; x *= 0x7feb352dU;
  x ^= x >> 15; x *= 0x846ca68bU;
  x ^= x >> 16; return x;
}
__device__ __forceinline__ float u01(uint32_t h) {          // (0,1]
  return ((h >> 8) + 1u) * (1.0f / 16777216.0f);
}
__device__ __forceinline__ float gauss(uint32_t c0) {
  float u1 = u01(hashu(c0));
  float u2 = u01(hashu(c0 + 1u));
  return sqrtf(-2.0f * logf(u1)) * __cosf(6.28318530717958647f * u2);
}

__device__ __forceinline__ float digammaf_dev(float x) {
  // asymptotic series, accurate for x > ~8 (here x >= 73.5)
  float inv = 1.0f / x, inv2 = inv * inv;
  return logf(x) - 0.5f * inv
       - inv2 * (1.0f/12.0f - inv2 * (1.0f/120.0f - inv2 * (1.0f/252.0f)));
}

// ---- CDNA5 async global->LDS copy (ASYNCcnt path, bypasses VGPRs) ----
__device__ __forceinline__ void async_copy_b128(uint32_t lds_off, const void* gaddr) {
  asm volatile("global_load_async_to_lds_b128 %0, %1, off"
               :: "v"(lds_off), "v"(gaddr) : "memory");
}
__device__ __forceinline__ void wait_async0() {
  asm volatile("s_wait_asynccnt 0x0" ::: "memory");
}

// ---------------- kernel 0: per-row stats (logZ, H, scale) ----------------
__global__ void stats_kernel(const float* __restrict__ scale1,
                             const float* __restrict__ scale2,
                             float* __restrict__ stats) {
  int r = blockIdx.x * blockDim.x + threadIdx.x;
  if (r >= N_ROWS) return;
  float scale = (r < 256) ? scale1[r] : scale2[r - 256];
  const float b = 63.5f;                 // (d-1)/2
  float a = b + scale;
  const float LN2 = 0.69314718055994531f;
  const float LNPI = 1.14472988584940017f;
  float logZ = -((a + b) * LN2 + lgammaf(a) - lgammaf(a + b) + b * LNPI);
  float H = -(logZ + scale * (LN2 + digammaf_dev(a) - digammaf_dev(a + b)));
  stats[r]            = logZ;
  stats[N_ROWS + r]   = H;
  stats[2*N_ROWS + r] = scale;
}

// ---------------- kernel 1: convert loc (concat) to f16 ----------------
__global__ void convert_kernel(const float* __restrict__ loc1,
                               const float* __restrict__ loc2,
                               _Float16* __restrict__ locH) {
  int idx = blockIdx.x * blockDim.x + threadIdx.x;
  if (idx >= N_ROWS * DIMS) return;
  int r = idx >> 7;
  float v = (r < 256) ? loc1[idx] : loc2[idx - 256 * DIMS];
  locH[idx] = (_Float16)v;
}

// ---------------- kernel 2: PowerSpherical rsample -> mc (f16) ----------------
__global__ __launch_bounds__(128)
void sample_kernel(const float* __restrict__ loc1, const float* __restrict__ loc2,
                   const float* __restrict__ scale1, const float* __restrict__ scale2,
                   _Float16* __restrict__ mc) {
  __shared__ float red[128];
  __shared__ float sc[1];
  const int row = blockIdx.x;          // 0..511
  const int s   = blockIdx.y;          // 0..999
  const int t   = threadIdx.x;         // 0..127 (dimension index)
  const float* locRow = (row < 256) ? (loc1 + row * DIMS) : (loc2 + (row - 256) * DIMS);
  const float scale   = (row < 256) ? scale1[row] : scale2[row - 256];
  const uint32_t base = ((uint32_t)s * 512u + (uint32_t)row) * 512u;

  // tangential gaussian v_t for dims 1..127
  float v = 0.0f;
  if (t >= 1) v = gauss(base + 2u * (uint32_t)t);
  red[t] = v * v; __syncthreads();
  #pragma unroll
  for (int off = 64; off > 0; off >>= 1) {
    if (t < off) red[t] += red[t + off];
    __syncthreads();
  }
  float vnorm = sqrtf(red[0]) + 1e-12f;
  __syncthreads();

  // marginal t: 2*Beta(a,b)-1 via normal approximation (a,b >= 63.5)
  if (t == 0) {
    const float b = 63.5f;
    float a = b + scale;
    float mean = a / (a + b);
    float var  = a * b / ((a + b) * (a + b) * (a + b + 1.0f));
    float g = gauss(base + 300u);
    float tv = 2.0f * (mean + sqrtf(var) * g) - 1.0f;
    sc[0] = fminf(fmaxf(tv, -0.999999f), 0.999999f);
  }
  __syncthreads();
  float tval = sc[0];
  float tang = sqrtf(fmaxf(1.0f - tval * tval, 1e-20f)) / vnorm;
  float y = (t == 0) ? tval : v * tang;

  // Householder e1 -> loc
  float uraw = ((t == 0) ? 1.0f : 0.0f) - locRow[t];
  red[t] = uraw * uraw; __syncthreads();
  #pragma unroll
  for (int off = 64; off > 0; off >>= 1) {
    if (t < off) red[t] += red[t + off];
    __syncthreads();
  }
  float u = uraw / (sqrtf(red[0]) + 1e-12f);
  __syncthreads();
  red[t] = y * u; __syncthreads();
  #pragma unroll
  for (int off = 64; off > 0; off >>= 1) {
    if (t < off) red[t] += red[t + off];
    __syncthreads();
  }
  float out = y - 2.0f * red[0] * u;
  mc[((size_t)s * N_ROWS + row) * DIMS + t] = (_Float16)out;
}

// ---------------- kernel 3: fused batched GEMM (WMMA f16) + log mean ----------------
// grid (8,8,SCHUNKS): 64x64 output tile; sample-chunk bz writes its own partial slice.
// B tiles double-buffered in LDS via async global->LDS copies (ASYNCcnt).
__global__ __launch_bounds__(256)
void gemm_log1p_kernel(const _Float16* __restrict__ locH,
                       const _Float16* __restrict__ mc,
                       float* __restrict__ meanPart) {
  __shared__ _Float16 Ah[64][136];      // loc tile, resident whole kernel
  __shared__ _Float16 Bh[2][64][136];   // double-buffered mc tiles

  const int tix = blockIdx.x;        // i tile (rows of loc)
  const int tij = blockIdx.y;        // j tile (rows of mc)
  const int chunk = blockIdx.z;
  const int t = threadIdx.x;
  const int lane = t & 31;
  const int wave = t >> 5;
  const int wm = wave & 3;           // 4 row groups of 16
  const int wn = wave >> 2;          // 2 col groups of 32

  // per-thread staging slots: 4 x 16B chunks of the 64x128 tile
  int srow[4], scol8[4];
  uint32_t ldsB[2][4];
  #pragma unroll
  for (int k = 0; k < 4; ++k) {
    int c = t + k * 256;
    srow[k] = c >> 4; scol8[k] = c & 15;
    ldsB[0][k] = (uint32_t)(uintptr_t)&Bh[0][srow[k]][scol8[k] * 8];
    ldsB[1][k] = (uint32_t)(uintptr_t)&Bh[1][srow[k]][scol8[k] * 8];
  }

  // stage A tile once (constant across all samples)
  #pragma unroll
  for (int k = 0; k < 4; ++k) {
    *(v8h*)&Ah[srow[k]][scol8[k] * 8] =
        *(const v8h*)&locH[((size_t)tix * 64 + srow[k]) * DIMS + scol8[k] * 8];
  }
  __syncthreads();

  // A fragments hoisted out of the sample loop (ISA 16-bit A layout:
  // lane m = lane&15, K-half select = (lane>>4)*8, second 8 halfs at +16)
  const int am  = wm * 16 + (lane & 15);
  const int ak0 = (lane >> 4) * 8;
  Frag afrag[4];
  #pragma unroll
  for (int kk = 0; kk < 4; ++kk) {
    afrag[kk].h[0] = *(const v8h*)&Ah[am][kk * 32 + ak0];
    afrag[kk].h[1] = *(const v8h*)&Ah[am][kk * 32 + ak0 + 16];
  }

  float acc[2][8];
  #pragma unroll
  for (int c = 0; c < 2; ++c)
    #pragma unroll
    for (int i = 0; i < 8; ++i) acc[c][i] = 0.0f;

  const int s0 = chunk * SPERCH;
  const _Float16* tileBase = mc + (size_t)tij * 64 * DIMS;

  // prime the pipeline: async-stage first sample into buffer 0
  {
    const _Float16* src = tileBase + (size_t)s0 * N_ROWS * DIMS;
    #pragma unroll
    for (int k = 0; k < 4; ++k)
      async_copy_b128(ldsB[0][k], src + srow[k] * DIMS + scol8[k] * 8);
  }

  const float LN2 = 0.69314718055994531f;
  int buf = 0;
  for (int s = s0; s < s0 + SPERCH; ++s) {
    wait_async0();                 // my async writes for Bh[buf] complete
    __syncthreads();               // everyone's complete; prev buffer free

    if (s + 1 < s0 + SPERCH) {     // overlap: stage next sample into other buffer
      const _Float16* nsrc = tileBase + (size_t)(s + 1) * N_ROWS * DIMS;
      #pragma unroll
      for (int k = 0; k < 4; ++k)
        async_copy_b128(ldsB[buf ^ 1][k], nsrc + srow[k] * DIMS + scol8[k] * 8);
    }

    #pragma unroll
    for (int c = 0; c < 2; ++c) {
      const int bn = (wn * 2 + c) * 16 + (lane & 15);  // mc row = WMMA B column
      v8f cacc = {};
      #pragma unroll
      for (int kk = 0; kk < 4; ++kk) {
        Frag bf;
        bf.h[0] = *(const v8h*)&Bh[buf][bn][kk * 32 + ak0];
        bf.h[1] = *(const v8h*)&Bh[buf][bn][kk * 32 + ak0 + 16];
        cacc = __builtin_amdgcn_wmma_f32_16x16x32_f16(
            false, afrag[kk].v, false, bf.v, (short)0, cacc, false, false);
      }
      // log1p(m), m in [-1,1]: bare v_log_f32 (no denorm guard needed for 1+m),
      // ln2 conversion folded into the accumulate FMA. TRANS co-executes with XDL.
      #pragma unroll
      for (int i = 0; i < 8; ++i)
        acc[c][i] = fmaf(__builtin_amdgcn_logf(1.0f + cacc[i]), LN2, acc[c][i]);
    }
    buf ^= 1;
  }

  // write partial mean slice (C/D layout: M = vgpr + 8*(lane>>4), N = lane&15)
  const float inv = 1.0f / (float)N_MC;
  const int rbase = tix * 64 + wm * 16 + 8 * (lane >> 4);
  float* dst = meanPart + (size_t)chunk * N_ROWS * N_ROWS;
  #pragma unroll
  for (int c = 0; c < 2; ++c) {
    const int col = tij * 64 + (wn * 2 + c) * 16 + (lane & 15);
    #pragma unroll
    for (int i = 0; i < 8; ++i)
      dst[(size_t)(rbase + i) * N_ROWS + col] = acc[c][i] * inv;
  }
}

// ---------------- kernel 4: per-row InfoNCE-style loss ----------------
__global__ __launch_bounds__(512)
void loss_kernel(const float* __restrict__ meanPart,
                 const float* __restrict__ stats,
                 float* __restrict__ rowLoss) {
  __shared__ float sh[512];
  const int i = blockIdx.x, j = threadIdx.x;
  float ml = 0.0f;
  #pragma unroll
  for (int c = 0; c < SCHUNKS; ++c)
    ml += meanPart[(size_t)c * N_ROWS * N_ROWS + (size_t)i * N_ROWS + j];
  float sim = -stats[N_ROWS + j] - stats[j] - stats[2 * N_ROWS + j] * ml;
  if (j == i) sim = -9e15f;
  sim *= 0.1f;           // TEMPERATURE

  sh[j] = sim; __syncthreads();
  #pragma unroll
  for (int off = 256; off > 0; off >>= 1) {
    if (j < off) sh[j] = fmaxf(sh[j], sh[j + off]);
    __syncthreads();
  }
  float mx = sh[0]; __syncthreads();
  sh[j] = __expf(sim - mx); __syncthreads();
  #pragma unroll
  for (int off = 256; off > 0; off >>= 1) {
    if (j < off) sh[j] += sh[j + off];
    __syncthreads();
  }
  float lse = mx + logf(sh[0]);
  __syncthreads();
  if (j == ((i + 256) & 511)) sh[0] = sim;   // positive pair
  __syncthreads();
  if (j == 0) rowLoss[i] = sh[0] - lse;
}

__global__ __launch_bounds__(512)
void final_kernel(const float* __restrict__ rowLoss, float* __restrict__ out) {
  __shared__ float sh[512];
  const int j = threadIdx.x;
  sh[j] = rowLoss[j]; __syncthreads();
  #pragma unroll
  for (int off = 256; off > 0; off >>= 1) {
    if (j < off) sh[j] += sh[j + off];
    __syncthreads();
  }
  if (j == 0) out[0] = sh[0] / 512.0f;
}

extern "C" void kernel_launch(void* const* d_in, const int* in_sizes, int n_in,
                              void* d_out, int out_size, void* d_ws, size_t ws_size,
                              hipStream_t stream) {
  (void)in_sizes; (void)n_in; (void)out_size; (void)ws_size;
  const float* loc1   = (const float*)d_in[0];
  const float* scale1 = (const float*)d_in[1];
  const float* loc2   = (const float*)d_in[2];
  const float* scale2 = (const float*)d_in[3];
  float* out = (float*)d_out;

  // workspace layout (16B-aligned slices)
  char* ws = (char*)d_ws;
  _Float16* locH   = (_Float16*)ws;                              // 512*128*2   = 128 KiB
  float*    stats  = (float*)(ws + 131072);                      // 3*512*4 (pad to 8 KiB)
  float*    meanP  = (float*)(ws + 131072 + 8192);               // 8*512*512*4 = 8 MiB
  float*    rowL   = (float*)(ws + 131072 + 8192 + 8388608);     // 512*4 (pad to 8 KiB)
  _Float16* mc     = (_Float16*)(ws + 131072 + 8192 + 8388608 + 8192); // 1000*512*128*2 = 125 MiB (L2-resident)

  convert_kernel<<<dim3((N_ROWS * DIMS + 255) / 256), dim3(256), 0, stream>>>(loc1, loc2, locH);
  stats_kernel<<<dim3(2), dim3(256), 0, stream>>>(scale1, scale2, stats);
  sample_kernel<<<dim3(N_ROWS, N_MC), dim3(128), 0, stream>>>(loc1, loc2, scale1, scale2, mc);
  gemm_log1p_kernel<<<dim3(8, 8, SCHUNKS), dim3(256), 0, stream>>>(locH, mc, meanP);
  loss_kernel<<<dim3(N_ROWS), dim3(512), 0, stream>>>(meanP, stats, rowL);
  final_kernel<<<dim3(1), dim3(512), 0, stream>>>(rowL, out);
}
